// PoolingLSTM_61795989454958
// MI455X (gfx1250) — matrix operand order repaired and checked by reading, not compiled, not run on hardware
//
#include <hip/hip_runtime.h>
#include <hip/hip_bf16.h>

typedef __bf16 bf16_t;
typedef __attribute__((ext_vector_type(16))) __bf16 bf16x16;
typedef __attribute__((ext_vector_type(8)))  float  f32x8;

namespace {
constexpr int N  = 1024;   // batch
constexpr int L  = 256;    // time steps
constexpr int H1 = 256;    // lstm1 hidden
constexpr int H2 = 512;    // lstm2 hidden
constexpr int MT = 16;     // batch rows per workgroup (one WMMA M-tile)
constexpr int NT = 256;    // threads per workgroup (8 waves)
}

__device__ __forceinline__ f32x8 wmma_bf16(bf16x16 a, bf16x16 b, f32x8 c) {
  // D = A(16x32) * B(32x16) + C, f32 accumulate
  return __builtin_amdgcn_wmma_f32_16x16x32_bf16(
      /*neg_a=*/false, a, /*neg_b=*/false, b, /*c_mod=*/(short)0, c,
      /*reuse_a=*/false, /*reuse_b=*/false);
}

__device__ __forceinline__ float frcp(float x) {
#if defined(__has_builtin)
#if __has_builtin(__builtin_amdgcn_rcpf)
  return __builtin_amdgcn_rcpf(x);    // v_rcp_f32, no IEEE div expansion
#else
  return 1.0f / x;
#endif
#else
  return 1.0f / x;
#endif
}

__device__ __forceinline__ float fsig(float x) { return frcp(1.0f + __expf(-x)); }
// tanh(x) = 2*sigmoid(2x) - 1 : correct saturation, avoids the ocml tanh libcall.
__device__ __forceinline__ float ftanh(float x) { return 2.0f * fsig(2.0f * x) - 1.0f; }

// Accumulate 4 gate tiles: acc[g] += A(16xK) * W[g*H+ctB .. +16, 0..K)^T.
// K-loop kept rolled (prevents load hoisting/spills); loads go straight into the
// WMMA-source unions (no rotation copies); addresses are incrementing pointers.
__device__ __forceinline__ void gemm_gates(const bf16_t* Alds, int lda, int K,
                                           const bf16_t* __restrict__ W, int ldw,
                                           int H, int ctB, int lane, f32x8 acc[4]) {
  const int mrow = lane & 15;
  const bf16_t* pa  = Alds + mrow * lda + (lane & 16);          // +16 elems upper half
  const int     bofs = lane & 16;                               // +16 elems upper half
  const bf16_t* pb0 = W + (size_t)(0 * H + ctB + mrow) * ldw + bofs;
  const bf16_t* pb1 = W + (size_t)(1 * H + ctB + mrow) * ldw + bofs;
  const bf16_t* pb2 = W + (size_t)(2 * H + ctB + mrow) * ldw + bofs;
  const bf16_t* pb3 = W + (size_t)(3 * H + ctB + mrow) * ldw + bofs;
#pragma clang loop unroll(disable)
  for (int kk = 0; kk < K; kk += 32) {
    union { bf16x16 v; uint4 u[2]; } a, b0, b1, b2, b3;
    a.u[0]  = *(const uint4*)(pa);        // ds_load_b128 x2
    a.u[1]  = *(const uint4*)(pa + 8);
    b0.u[0] = *(const uint4*)(pb0);       // global_load_b128 x8 (one clause)
    b0.u[1] = *(const uint4*)(pb0 + 8);
    b1.u[0] = *(const uint4*)(pb1);
    b1.u[1] = *(const uint4*)(pb1 + 8);
    b2.u[0] = *(const uint4*)(pb2);
    b2.u[1] = *(const uint4*)(pb2 + 8);
    b3.u[0] = *(const uint4*)(pb3);
    b3.u[1] = *(const uint4*)(pb3 + 8);
    acc[0] = wmma_bf16(a.v, b0.v, acc[0]);
    acc[1] = wmma_bf16(a.v, b1.v, acc[1]);
    acc[2] = wmma_bf16(a.v, b2.v, acc[2]);
    acc[3] = wmma_bf16(a.v, b3.v, acc[3]);
    pa += 32; pb0 += 32; pb1 += 32; pb2 += 32; pb3 += 32;
  }
}

__global__ void cvt_bf16_kernel(const float* __restrict__ src,
                                bf16_t* __restrict__ dst, int n) {
  int i = blockIdx.x * blockDim.x + threadIdx.x;
  if (i < n) dst[i] = (bf16_t)src[i];
}

__global__ void __launch_bounds__(NT, 1)
lstm_pool_kernel(const float* __restrict__ x,
                 const float* __restrict__ h0_1, const float* __restrict__ c0_1,
                 const float* __restrict__ h0_2, const float* __restrict__ c0_2,
                 const float* __restrict__ W_ih1,
                 const float* __restrict__ b_ih1, const float* __restrict__ b_hh1,
                 const float* __restrict__ b_ih2, const float* __restrict__ b_hh2,
                 const float* __restrict__ W_lin, const float* __restrict__ b_lin,
                 const float* __restrict__ W_pool, const float* __restrict__ b_pool,
                 const bf16_t* __restrict__ Whh1b,
                 const bf16_t* __restrict__ Wih2b,
                 const bf16_t* __restrict__ Whh2b,
                 float* __restrict__ out) {
  // Ping-pong hidden-state buffers (removes WAR hazards inside a step).
  __shared__ __align__(16) bf16_t h1_lds[2][MT * H1];   // 16 KB
  __shared__ __align__(16) bf16_t h2_lds[2][MT * H2];   // 32 KB
  __shared__ __align__(16) float  x_lds[MT * L];        // 16 KB
  __shared__ float b1_lds[4 * H1], wih1_lds[4 * H1];    // 8 KB
  __shared__ float b2_lds[4 * H2];                      // 8 KB
  __shared__ __align__(16) float wlin_lds[H2];          // 2 KB
  __shared__ float wpool_lds[L];                        // 1 KB
  __shared__ float ypart[NT];
  __shared__ float pred_lds[MT];

  const int tid  = threadIdx.x;
  const int lane = tid & 31;
  const int wv   = tid >> 5;            // wave 0..7
  const int hi8  = (lane & 16) >> 1;    // +8 row offset for upper half-wave (C/D layout)
  const int m0   = blockIdx.x * MT;     // batch-row base of this workgroup
  const float blin  = b_lin[0];
  const float bpool = b_pool[0];

  // ---------------- one-time LDS preload ----------------
  for (int i = tid; i < 4 * H1; i += NT) { b1_lds[i] = b_ih1[i] + b_hh1[i]; wih1_lds[i] = W_ih1[i]; }
  for (int i = tid; i < 4 * H2; i += NT) b2_lds[i] = b_ih2[i] + b_hh2[i];
  for (int i = tid; i < H2; i += NT) wlin_lds[i] = W_lin[i];
  for (int i = tid; i < L;  i += NT) wpool_lds[i] = W_pool[i];
  if (tid < MT) pred_lds[tid] = 0.0f;
  for (int i = tid; i < MT * L;  i += NT) x_lds[i]     = x[(size_t)(m0 + i / L) * L + (i % L)];
  for (int i = tid; i < MT * H1; i += NT) h1_lds[0][i] = (bf16_t)h0_1[(size_t)(m0 + i / H1) * H1 + (i % H1)];
  for (int i = tid; i < MT * H2; i += NT) h2_lds[0][i] = (bf16_t)h0_2[(size_t)(m0 + i / H2) * H2 + (i % H2)];

  // Cell state in registers, laid out exactly like the WMMA C/D fragment:
  // lane (&15) = column within tile, VGPR r = row (r + hi8).
  float c1st[2][8], c2st[4][8];
#pragma unroll
  for (int ct = 0; ct < 2; ++ct) {
    const int j = (2 * wv + ct) * 16 + (lane & 15);
#pragma unroll
    for (int r = 0; r < 8; ++r) c1st[ct][r] = c0_1[(size_t)(m0 + r + hi8) * H1 + j];
  }
#pragma unroll
  for (int ct = 0; ct < 4; ++ct) {
    const int j = (4 * wv + ct) * 16 + (lane & 15);
#pragma unroll
    for (int r = 0; r < 8; ++r) c2st[ct][r] = c0_2[(size_t)(m0 + r + hi8) * H2 + j];
  }
  __syncthreads();

  // ---------------- time recurrence ----------------
  for (int t = 0; t < L; ++t) {
    const bf16_t* h1r = h1_lds[t & 1];
    bf16_t*       h1w = h1_lds[(t & 1) ^ 1];
    const bf16_t* h2r = h2_lds[t & 1];
    bf16_t*       h2w = h2_lds[(t & 1) ^ 1];

    // ===== LSTM cell 1: z1 = h1 @ Whh1^T + x_t * Wih1 + (b_ih1+b_hh1) =====
#pragma unroll
    for (int ct = 0; ct < 2; ++ct) {
      const int ctB = (2 * wv + ct) * 16;
      const int j   = ctB + (lane & 15);
      f32x8 acc[4];
#pragma unroll
      for (int g = 0; g < 4; ++g) {
        const float bg = b1_lds[g * H1 + j];
        const float wg = wih1_lds[g * H1 + j];
#pragma unroll
        for (int r = 0; r < 8; ++r)
          acc[g][r] = bg + x_lds[(r + hi8) * L + t] * wg;
      }
      gemm_gates(h1r, H1, H1, Whh1b, H1, H1, ctB, lane, acc);
#pragma unroll
      for (int r = 0; r < 8; ++r) {               // i,f,g,o gate order (PyTorch)
        const float ig = fsig(acc[0][r]);
        const float fg = fsig(acc[1][r]);
        const float gg = ftanh(acc[2][r]);
        const float og = fsig(acc[3][r]);
        const float c  = fg * c1st[ct][r] + ig * gg;
        c1st[ct][r] = c;
        h1w[(r + hi8) * H1 + j] = (bf16_t)(og * ftanh(c));
      }
    }
    __syncthreads();   // h1_t visible to all waves

    // ===== LSTM cell 2: z2 = h1_t @ Wih2^T + h2 @ Whh2^T + (b_ih2+b_hh2) =====
#pragma unroll
    for (int ct = 0; ct < 4; ++ct) {
      const int ctB = (4 * wv + ct) * 16;
      const int j   = ctB + (lane & 15);
      f32x8 acc[4];
#pragma unroll
      for (int g = 0; g < 4; ++g) {
        const float bg = b2_lds[g * H2 + j];
#pragma unroll
        for (int r = 0; r < 8; ++r) acc[g][r] = bg;
      }
      gemm_gates(h1w, H1, H1, Wih2b, H1, H2, ctB, lane, acc);
      gemm_gates(h2r, H2, H2, Whh2b, H2, H2, ctB, lane, acc);
#pragma unroll
      for (int r = 0; r < 8; ++r) {
        const float ig = fsig(acc[0][r]);
        const float fg = fsig(acc[1][r]);
        const float gg = ftanh(acc[2][r]);
        const float og = fsig(acc[3][r]);
        const float c  = fg * c2st[ct][r] + ig * gg;
        c2st[ct][r] = c;
        h2w[(r + hi8) * H2 + j] = (bf16_t)(og * ftanh(c));
      }
    }
    __syncthreads();   // h2_t visible

    // ===== y_t = h2_t @ W_lin^T + b_lin, folded into pred += W_pool[t] * y_t =====
    {
      const int m  = tid >> 4;       // row 0..15
      const int cc = tid & 15;       // 16 threads per row, 32 columns each
      const bf16_t* row = h2w + m * H2 + cc * 32;   // 64B aligned
      float s = 0.0f;
#pragma unroll
      for (int v = 0; v < 4; ++v) {
        union { uint4 u; __bf16 h[8]; } hv;
        hv.u = *(const uint4*)(row + v * 8);        // ds_load_b128
        const float4 w0 = *(const float4*)(&wlin_lds[cc * 32 + v * 8]);
        const float4 w1 = *(const float4*)(&wlin_lds[cc * 32 + v * 8 + 4]);
        s += (float)hv.h[0] * w0.x + (float)hv.h[1] * w0.y +
             (float)hv.h[2] * w0.z + (float)hv.h[3] * w0.w +
             (float)hv.h[4] * w1.x + (float)hv.h[5] * w1.y +
             (float)hv.h[6] * w1.z + (float)hv.h[7] * w1.w;
      }
      ypart[tid] = s;
    }
    __syncthreads();
    if (tid < MT) {
      float tot = blin;
      for (int c2 = 0; c2 < 16; ++c2) tot += ypart[tid * 16 + c2];
      pred_lds[tid] += wpool_lds[t] * tot;
    }
    // no trailing barrier needed: ypart is next written only after two barriers
  }

  __syncthreads();
  if (tid < MT) out[m0 + tid] = pred_lds[tid] + bpool;
}

extern "C" void kernel_launch(void* const* d_in, const int* in_sizes, int n_in,
                              void* d_out, int out_size, void* d_ws, size_t ws_size,
                              hipStream_t stream) {
  const float* x      = (const float*)d_in[0];
  const float* h0_1   = (const float*)d_in[1];
  const float* c0_1   = (const float*)d_in[2];
  const float* h0_2   = (const float*)d_in[3];
  const float* c0_2   = (const float*)d_in[4];
  const float* W_ih1  = (const float*)d_in[5];
  const float* W_hh1  = (const float*)d_in[6];
  const float* b_ih1  = (const float*)d_in[7];
  const float* b_hh1  = (const float*)d_in[8];
  const float* W_ih2  = (const float*)d_in[9];
  const float* W_hh2  = (const float*)d_in[10];
  const float* b_ih2  = (const float*)d_in[11];
  const float* b_hh2  = (const float*)d_in[12];
  const float* W_lin  = (const float*)d_in[13];
  const float* b_lin  = (const float*)d_in[14];
  const float* W_pool = (const float*)d_in[15];
  const float* b_pool = (const float*)d_in[16];

  // Workspace: bf16 copies of the three big weight matrices (3.5 MB total).
  bf16_t* Whh1b = (bf16_t*)d_ws;
  bf16_t* Wih2b = Whh1b + 4 * H1 * H1;
  bf16_t* Whh2b = Wih2b + 4 * H2 * H1;

  const int n1 = 4 * H1 * H1;   // 262144
  const int n2 = 4 * H2 * H1;   // 524288
  const int n3 = 4 * H2 * H2;   // 1048576
  cvt_bf16_kernel<<<(n1 + 255) / 256, 256, 0, stream>>>(W_hh1, Whh1b, n1);
  cvt_bf16_kernel<<<(n2 + 255) / 256, 256, 0, stream>>>(W_ih2, Wih2b, n2);
  cvt_bf16_kernel<<<(n3 + 255) / 256, 256, 0, stream>>>(W_hh2, Whh2b, n3);

  lstm_pool_kernel<<<dim3(N / MT), dim3(NT), 0, stream>>>(
      x, h0_1, c0_1, h0_2, c0_2, W_ih1, b_ih1, b_hh1, b_ih2, b_hh2,
      W_lin, b_lin, W_pool, b_pool, Whh1b, Wih2b, Whh2b, (float*)d_out);
}
